// EI_Structure_Feature_22445499089564
// MI455X (gfx1250) — matrix operand order repaired and checked by reading, not compiled, not run on hardware
//
#include <hip/hip_runtime.h>

typedef __attribute__((ext_vector_type(16))) _Float16 v16h;
typedef __attribute__((ext_vector_type(8)))  _Float16 h8;
typedef __attribute__((ext_vector_type(4)))  _Float16 h4;
typedef __attribute__((ext_vector_type(8)))  float    v8f;

#define BATCH 4
#define CH    128
#define HH    192
#define WW    384
#define HW    (HH * WW)            // 73728
#define NPIX  (BATCH * HW)         // 294912
#define KTOT  256
#define NF    ((float)NPIX)
#define EPS_BN 1e-5f

// ---------------------------------------------------------------- dot map ---
// dot[p] = <img_n, evt_n> over channels, with L2 normalization folded in.
__global__ __launch_bounds__(256)
void ei_dot_kernel(const float* __restrict__ img, const float* __restrict__ evt,
                   float* __restrict__ dotm) {
  int p = blockIdx.x * 256 + threadIdx.x;            // exact: 1152*256 == NPIX
  int b = p / HW;
  int hw = p - b * HW;
  const float* ip = img + (long long)b * CH * HW + hw;
  const float* ep = evt + (long long)b * CH * HW + hw;
  float sii = 0.f, see = 0.f, sie = 0.f;
#pragma unroll 4
  for (int c = 0; c < CH; ++c) {
    float a = ip[(long long)c * HW];
    float e = ep[(long long)c * HW];
    sii += a * a; see += e * e; sie += a * e;
  }
  float ni = sqrtf(sii); ni = ni > 1e-12f ? ni : 1e-12f;
  float ne = sqrtf(see); ne = ne > 1e-12f ? ne : 1e-12f;
  dotm[p] = sie / (ni * ne);
}

// ---------------------------------------------- affinity moment reduction ---
// Per dilation: M[9] = sum aff_c,  S[45] = sum aff_c*aff_c' (upper triangle).
__global__ __launch_bounds__(256)
void ei_affstats_kernel(const float* __restrict__ dotm, float* __restrict__ stats1) {
  int d = 1 << blockIdx.y;                            // dilations 1,2,4,8
  int p = blockIdx.x * 256 + threadIdx.x;
  int b = p / HW;
  int hw = p - b * HW;
  int h = hw / WW, w = hw - h * WW;

  float aff[9];
  int k = 0;
#pragma unroll
  for (int ky = 0; ky < 3; ++ky)
#pragma unroll
    for (int kx = 0; kx < 3; ++kx) {
      int hh = h + (ky - 1) * d, ww2 = w + (kx - 1) * d;
      float v = 0.f;
      if (hh >= 0 && hh < HH && ww2 >= 0 && ww2 < WW)
        v = dotm[b * HW + hh * WW + ww2];
      aff[k++] = v > 0.f ? v : 0.f;
    }

  __shared__ float red[54];
  if (threadIdx.x < 54) red[threadIdx.x] = 0.f;
  __syncthreads();
  int si = 9;
#pragma unroll
  for (int c = 0; c < 9; ++c) {
    atomicAdd(&red[c], aff[c]);
#pragma unroll
    for (int c2 = c; c2 < 9; ++c2)
      atomicAdd(&red[si++], aff[c] * aff[c2]);
  }
  __syncthreads();
  if (threadIdx.x < 54)
    atomicAdd(&stats1[blockIdx.y * 54 + threadIdx.x], red[threadIdx.x]);
}

// --------------------------------------------- fold BN1 into scale / bias ---
__global__ void ei_fold1_kernel(const float* w1, const float* g1, const float* b1,
                                const float* w2, const float* g2, const float* b2,
                                const float* w3, const float* g3, const float* b3,
                                const float* w4, const float* g4, const float* b4,
                                const float* __restrict__ stats1,
                                float* __restrict__ coef1) {
  int t = threadIdx.x;
  if (t >= 128) return;
  int dil = t >> 5, o = t & 31;
  const float *wp, *gp, *bp;
  if (dil == 0)      { wp = w1; gp = g1; bp = b1; }
  else if (dil == 1) { wp = w2; gp = g2; bp = b2; }
  else if (dil == 2) { wp = w3; gp = g3; bp = b3; }
  else               { wp = w4; gp = g4; bp = b4; }
  const float* st = stats1 + dil * 54;
  float wr[9];
#pragma unroll
  for (int c = 0; c < 9; ++c) wr[c] = wp[o * 9 + c];
  float mean = 0.f;
#pragma unroll
  for (int c = 0; c < 9; ++c) mean += wr[c] * st[c];
  mean /= NF;
  float q = 0.f;
  int si = 9;
#pragma unroll
  for (int c = 0; c < 9; ++c)
#pragma unroll
    for (int c2 = c; c2 < 9; ++c2) {
      float s = st[si++];
      q += wr[c] * wr[c2] * s * (c == c2 ? 1.f : 2.f);
    }
  q /= NF;
  float var = q - mean * mean;
  float a = gp[o] * rsqrtf(var + EPS_BN);
  coef1[t * 2]     = a;
  coef1[t * 2 + 1] = bp[o] - a * mean;
}

// ----------------------------------- fused affinity -> conv1 -> BN -> f16 ---
// One wave per pixel: lane l computes output channels 4l..4l+3 (dil = l/8),
// writes pixel-major f16 feats[p][128] fully coalesced (8B per lane).
__global__ __launch_bounds__(256)
void ei_feats_kernel(const float* __restrict__ dotm,
                     const float* __restrict__ w1, const float* __restrict__ w2,
                     const float* __restrict__ w3, const float* __restrict__ w4,
                     const float* __restrict__ coef1, _Float16* __restrict__ feats) {
  int lane = threadIdx.x & 31;
  int warp = threadIdx.x >> 5;
  int p = blockIdx.x * 8 + warp;                      // exact: 36864*8 == NPIX
  int b = p / HW;
  int hw = p - b * HW;
  int h = hw / WW, w = hw - h * WW;
  int dil = lane >> 3;
  int d = 1 << dil;
  const float* wp = dil == 0 ? w1 : dil == 1 ? w2 : dil == 2 ? w3 : w4;

  float aff[9];
  int k = 0;
#pragma unroll
  for (int ky = 0; ky < 3; ++ky)
#pragma unroll
    for (int kx = 0; kx < 3; ++kx) {
      int hh = h + (ky - 1) * d, ww2 = w + (kx - 1) * d;
      float v = 0.f;
      if (hh >= 0 && hh < HH && ww2 >= 0 && ww2 < WW)
        v = dotm[b * HW + hh * WW + ww2];
      aff[k++] = v > 0.f ? v : 0.f;
    }

  int obase = (lane & 7) * 4;
  h4 out;
#pragma unroll
  for (int j = 0; j < 4; ++j) {
    int o = obase + j;
    float v = 0.f;
#pragma unroll
    for (int c = 0; c < 9; ++c) v += wp[o * 9 + c] * aff[c];
    float a  = coef1[(dil * 32 + o) * 2];
    float cc = coef1[(dil * 32 + o) * 2 + 1];
    float f = a * v + cc;
    out[j] = (_Float16)(f > 0.f ? f : 0.f);
  }
  *(h4*)(feats + (long long)p * 128 + lane * 4) = out;
}

// ------------------------------------------------- weights fp32 -> f16 ------
__global__ __launch_bounds__(256)
void ei_wconv_kernel(const float* __restrict__ wimg, const float* __restrict__ wevt,
                     _Float16* __restrict__ wh) {
  int i = blockIdx.x * 256 + threadIdx.x;             // 256*256 == 65536 == 2*128*256/2... exact below
  if (i < CH * KTOT)            wh[i] = (_Float16)wimg[i];
  else if (i < 2 * CH * KTOT)   wh[i] = (_Float16)wevt[i - CH * KTOT];
}

// ------------------------------------------------------- WMMA big GEMM ------
// grid.y = conv (0: image-mix, 1: event-mix). One wave = 128 out-ch x 16 px.
__global__ __launch_bounds__(256)
void ei_gemm_kernel(const float* __restrict__ img, const float* __restrict__ evt,
                    const _Float16* __restrict__ wh, const _Float16* __restrict__ feats,
                    float* __restrict__ y, float* __restrict__ stats2) {
  const int lane = threadIdx.x & 31;
  const int wave = threadIdx.x >> 5;
  const int conv = blockIdx.y;
  const int tile = blockIdx.x * 8 + wave;             // exact: 2304*8 == NPIX/16
  const int p0 = tile * 16;
  const int col = p0 + (lane & 15);
  const int b = col / HW;                             // tiles never straddle batches
  const int hw = col - b * HW;
  const bool hi = lane >= 16;
  const float* src = conv ? evt : img;
  const _Float16* W = wh + (long long)conv * CH * KTOT;
  const long long inbase = (long long)b * CH * HW + hw;

  __shared__ float s_sum[128];
  __shared__ float s_sq[128];
  for (int i = threadIdx.x; i < 128; i += 256) { s_sum[i] = 0.f; s_sq[i] = 0.f; }
  __syncthreads();

  v8f acc[8] = {};
  const int kb = hi ? 16 : 0;                         // B frag: lane group -> K half

  for (int kk = 0; kk < 8; ++kk) {
    const int c0 = kk * 32;
    v16h bfrag;
    if (c0 < 128) {
      // fp32 inputs, convert in-register (channel-strided, coalesced across lanes)
      const float* s = src + inbase + (long long)(c0 + kb) * HW;
#pragma unroll
      for (int t = 0; t < 16; ++t) bfrag[t] = (_Float16)s[(long long)t * HW];
    } else {
      // pixel-major f16 feats: one 32B contiguous load per lane
      const h8* f8 = (const h8*)(feats + (long long)col * 128 + (c0 - 128) + kb);
      h8 lo = f8[0], hp = f8[1];
#pragma unroll
      for (int t = 0; t < 8; ++t) { bfrag[t] = lo[t]; bfrag[t + 8] = hp[t]; }
    }
#pragma unroll
    for (int ot = 0; ot < 8; ++ot) {
      // A frag: lanes<16 K=c0+{0..15}, lanes>=16 K=c0+{16..31}
      const _Float16* wr = W + (long long)(ot * 16 + (lane & 15)) * KTOT + c0 + (hi ? 16 : 0);
      h8 alo = *(const h8*)wr;
      h8 ahi = *(const h8*)(wr + 8);
      v16h afrag;
#pragma unroll
      for (int t = 0; t < 8; ++t) { afrag[t] = alo[t]; afrag[t + 8] = ahi[t]; }
      acc[ot] = __builtin_amdgcn_wmma_f32_16x16x32_f16(
          false, afrag, false, bfrag, (short)0, acc[ot], false, false);
    }
  }

  // Epilogue: write pre-BN y into d_out, accumulate per-channel moments.
  float* yout = y + (long long)conv * NPIX * CH + (long long)b * CH * HW + hw;
#pragma unroll
  for (int ot = 0; ot < 8; ++ot) {
#pragma unroll
    for (int r = 0; r < 8; ++r) {
      int ch = ot * 16 + r + (hi ? 8 : 0);
      float v = acc[ot][r];
      yout[(long long)ch * HW] = v;
      atomicAdd(&s_sum[ch], v);
      atomicAdd(&s_sq[ch], v * v);
    }
  }
  __syncthreads();
  if (threadIdx.x < 128) {
    atomicAdd(&stats2[conv * 256 + threadIdx.x], s_sum[threadIdx.x]);
    atomicAdd(&stats2[conv * 256 + 128 + threadIdx.x], s_sq[threadIdx.x]);
  }
}

// --------------------------------------------- fold BN2 into scale / bias ---
__global__ void ei_fold2_kernel(const float* gimg, const float* bimg,
                                const float* gevt, const float* bevt,
                                const float* __restrict__ stats2,
                                float* __restrict__ coef2) {
  int t = threadIdx.x;
  if (t >= 256) return;
  int conv = t >> 7, ch = t & 127;
  float sum = stats2[conv * 256 + ch];
  float sq  = stats2[conv * 256 + 128 + ch];
  float mean = sum / NF;
  float var  = sq / NF - mean * mean;
  const float* g = conv ? gevt : gimg;
  const float* bb = conv ? bevt : bimg;
  float a = g[ch] * rsqrtf(var + EPS_BN);
  coef2[t * 2]     = a;
  coef2[t * 2 + 1] = bb[ch] - a * mean;
}

// ----------------------------------------------- in-place BN + ReLU apply ---
__global__ __launch_bounds__(256)
void ei_bnapply_kernel(float* __restrict__ y, const float* __restrict__ coef2) {
  long long i = (long long)blockIdx.x * 256 + threadIdx.x;   // exact multiple
  long long conv = i / ((long long)NPIX * CH);
  long long rem = i - conv * ((long long)NPIX * CH);
  int ch = (int)((rem / HW) & 127);
  float a  = coef2[(conv * 128 + ch) * 2];
  float cc = coef2[(conv * 128 + ch) * 2 + 1];
  float v = a * y[i] + cc;
  y[i] = v > 0.f ? v : 0.f;
}

// ---------------------------------------------------------------- launch ----
extern "C" void kernel_launch(void* const* d_in, const int* in_sizes, int n_in,
                              void* d_out, int out_size, void* d_ws, size_t ws_size,
                              hipStream_t stream) {
  const float* image = (const float*)d_in[0];
  const float* event = (const float*)d_in[1];
  const float* w1 = (const float*)d_in[2];
  const float* g1 = (const float*)d_in[3];
  const float* b1 = (const float*)d_in[4];
  const float* w2 = (const float*)d_in[5];
  const float* g2 = (const float*)d_in[6];
  const float* b2 = (const float*)d_in[7];
  const float* w3 = (const float*)d_in[8];
  const float* g3 = (const float*)d_in[9];
  const float* b3 = (const float*)d_in[10];
  const float* w4 = (const float*)d_in[11];
  const float* g4 = (const float*)d_in[12];
  const float* b4 = (const float*)d_in[13];
  const float* w_img = (const float*)d_in[14];
  const float* g_img = (const float*)d_in[15];
  const float* b_img = (const float*)d_in[16];
  const float* w_evt = (const float*)d_in[17];
  const float* g_evt = (const float*)d_in[18];
  const float* b_evt = (const float*)d_in[19];
  float* out = (float*)d_out;

  char* ws = (char*)d_ws;
  float*    dotm   = (float*)   (ws);                 //  1,179,648 B
  _Float16* feats  = (_Float16*)(ws + 1179648);       // 75,497,472 B
  _Float16* wh     = (_Float16*)(ws + 76677120);      //    131,072 B
  float*    stats1 = (float*)   (ws + 76808192);      //      1,024 B (4*54 used)
  float*    stats2 = (float*)   (ws + 76809216);      //      2,048 B
  float*    coef1  = (float*)   (ws + 76811264);      //      1,024 B
  float*    coef2  = (float*)   (ws + 76812288);      //      2,048 B

  // zero the moment accumulators (graph-capture-safe)
  hipMemsetAsync(ws + 76808192, 0, 3072, stream);

  // fp32 -> f16 weight conversion for the big convs (2*128*256 = 65536 elems)
  ei_wconv_kernel<<<dim3(256), dim3(256), 0, stream>>>(w_img, w_evt, wh);

  // normalized dot map
  ei_dot_kernel<<<dim3(NPIX / 256), dim3(256), 0, stream>>>(image, event, dotm);

  // affinity moments per dilation
  ei_affstats_kernel<<<dim3(NPIX / 256, 4), dim3(256), 0, stream>>>(dotm, stats1);

  // fold BN of the 9->32 convs
  ei_fold1_kernel<<<dim3(1), dim3(128), 0, stream>>>(
      w1, g1, b1, w2, g2, b2, w3, g3, b3, w4, g4, b4, stats1, coef1);

  // fused affinity -> conv1 -> BN -> ReLU -> f16 feats (pixel-major)
  ei_feats_kernel<<<dim3(NPIX / 8), dim3(256), 0, stream>>>(
      dotm, w1, w2, w3, w4, coef1, feats);

  // WMMA GEMM: pre-BN outputs + per-channel moments
  ei_gemm_kernel<<<dim3(NPIX / 16 / 8, 2), dim3(256), 0, stream>>>(
      image, event, wh, feats, out, stats2);

  // fold BN of the 256->128 convs
  ei_fold2_kernel<<<dim3(1), dim3(256), 0, stream>>>(
      g_img, b_img, g_evt, b_evt, stats2, coef2);

  // in-place BN + ReLU
  ei_bnapply_kernel<<<dim3((unsigned)((2LL * NPIX * CH) / 256)), dim3(256), 0, stream>>>(
      out, coef2);
}
